// ConcatCritic_13357348291414
// MI455X (gfx1250) — compile-verified, matrix operand order checked by reading, not run on hardware
//
#include <hip/hip_runtime.h>

// ---------------------------------------------------------------------------
// ConcatCritic fused kernel for gfx1250 (MI455X).
// out[i,j] = W3 . relu(W2 . relu(hx[i]+hy[j]+b1) + b2) + b3,  i,j in [0,512)
// hx = x @ W1[:, :128]^T, hy = y @ W1[:, 128:]^T
// Heavy GEMM (512^2 pairs x 512 x 512) runs on v_wmma_f32_16x16x32_f16.
// ---------------------------------------------------------------------------

typedef __attribute__((ext_vector_type(16))) _Float16     v16h;
typedef __attribute__((ext_vector_type(8)))  float        v8f;
typedef __attribute__((ext_vector_type(8)))  unsigned int v8u;

#define BSZ   512   // batch (i and j)
#define HID   512   // hidden width H
#define DIN   128   // per-input feature dim
#define ROWP  520   // padded LDS row stride in halves (512 + 8 -> bank-conflict-free)
#define PAIRS 128   // pairs per workgroup: 16 i  x  8 j
#define NTILE 16    // WMMA N tile
#define KSTEP 32    // WMMA K per instruction

// -------------------- prep: hx, hy (fp32) + W2 -> f16 ----------------------
__global__ __launch_bounds__(256) void cc_prep(const float* __restrict__ x,
                                               const float* __restrict__ y,
                                               const float* __restrict__ W1,
                                               const float* __restrict__ W2,
                                               float* __restrict__ hx,
                                               float* __restrict__ hy,
                                               _Float16* __restrict__ W2h) {
  int gid = blockIdx.x * 256 + threadIdx.x;
  if (blockIdx.x < 2048) {                    // hx then hy: 2 * 512*512 dots of length 128
    int sel = (gid >= BSZ * HID) ? 1 : 0;     // 0 -> hx, 1 -> hy
    int idx = gid - sel * (BSZ * HID);        // [0, 262144)
    int row = idx >> 9;                       // i (or j)
    int h   = idx & (HID - 1);
    const float* vec = sel ? (y + row * DIN) : (x + row * DIN);
    const float* w   = W1 + h * (2 * DIN) + sel * DIN;   // W1 row-major [512][256]
    float s = 0.f;
#pragma unroll 8
    for (int d = 0; d < DIN; ++d) s += vec[d] * w[d];
    (sel ? hy : hx)[idx] = s;
  } else {                                    // W2 fp32 -> f16 copy (row-major [512][512])
    int idx = gid - 2048 * 256;               // [0, 262144)
    W2h[idx] = (_Float16)W2[idx];
  }
}

// Load one 16x32 f16 fragment row from LDS per the documented gfx1250 layout:
// lanes 0-15 hold K = kb+{0..7, 16..23}; lanes 16-31 hold K = kb+{8..15, 24..31}.
__device__ __forceinline__ v16h cc_frag(const _Float16* row, int hi, int kb) {
  const uint4 q0 = *(const uint4*)(row + kb + hi * 8);
  const uint4 q1 = *(const uint4*)(row + kb + hi * 8 + 16);
  v8u r;
  r[0] = q0.x; r[1] = q0.y; r[2] = q0.z; r[3] = q0.w;
  r[4] = q1.x; r[5] = q1.y; r[6] = q1.z; r[7] = q1.w;
  return __builtin_bit_cast(v16h, r);
}

// ----------------------------- main kernel ---------------------------------
__global__ __launch_bounds__(256) void cc_main(const float* __restrict__ hx,
                                               const float* __restrict__ hy,
                                               const float* __restrict__ b1,
                                               const _Float16* __restrict__ W2h,
                                               const float* __restrict__ b2,
                                               const float* __restrict__ W3,
                                               const float* __restrict__ b3,
                                               float* __restrict__ out) {
  __shared__ __align__(16) _Float16 vbuf[PAIRS * ROWP];  // ~130 KB: relu(hx+hy+b1) in f16
  __shared__ __align__(16) _Float16 bbuf[NTILE * ROWP];  // ~16 KB : W2 strip (16 rows)

  const int tid  = threadIdx.x;
  const int lane = tid & 31;
  const int wave = tid >> 5;           // 8 waves; wave w owns pairs [16w, 16w+16)
  const int col  = lane & 15;
  const int hi   = lane >> 4;
  const int i0   = blockIdx.x * 16;    // 32 i-tiles
  const int j0   = blockIdx.y * 8;     // 64 j-tiles

  // ---- stage v[pair][h] = relu(hx[i,h] + hy[j,h] + b1[h]) as f16 in LDS ----
  for (int idx = tid; idx < PAIRS * HID; idx += 256) {
    int p = idx >> 9;                  // pair 0..127
    int h = idx & (HID - 1);
    int il = p >> 3, jl = p & 7;       // pair = i_local*8 + j_local
    float v = hx[(i0 + il) * HID + h] + hy[(j0 + jl) * HID + h] + b1[h];
    v = v > 0.f ? v : 0.f;
    vbuf[p * ROWP + h] = (_Float16)v;
  }

  float acc[8];
#pragma unroll
  for (int r = 0; r < 8; ++r) acc[r] = 0.f;

  const _Float16* arow = vbuf + (wave * 16 + col) * ROWP;  // A: this lane's pair row
  const _Float16* brow = bbuf + col * ROWP;                // B: this lane's W2 row (n)

  for (int nt = 0; nt < 32; ++nt) {
    const int nb = nt * NTILE;
    __syncthreads();                   // staging done (nt==0) / bbuf consumers done (nt>0)
    {                                  // cooperative load: W2h rows nb..nb+15 (16 KB)
      int r = tid >> 4;                // 0..15 row
      int c = (tid & 15) * 32;         // 32 halves per thread
      const _Float16* src = W2h + (size_t)(nb + r) * HID + c;
      _Float16*       dst = bbuf + r * ROWP + c;
      *(uint4*)(dst)      = *(const uint4*)(src);
      *(uint4*)(dst + 8)  = *(const uint4*)(src + 8);
      *(uint4*)(dst + 16) = *(const uint4*)(src + 16);
      *(uint4*)(dst + 24) = *(const uint4*)(src + 24);
      if (nt < 31)                     // gfx1250 global_prefetch_b8 of next strip
        __builtin_prefetch((const void*)(W2h + (size_t)(nb + NTILE + r) * HID + c), 0, 0);
    }
    __syncthreads();

    v8f d = {0.f, 0.f, 0.f, 0.f, 0.f, 0.f, 0.f, 0.f};
#pragma unroll
    for (int ks = 0; ks < HID / KSTEP; ++ks) {   // K = 512 in steps of 32
      v16h a = cc_frag(arow, hi, ks * KSTEP);
      v16h b = cc_frag(brow, hi, ks * KSTEP);
      d = __builtin_amdgcn_wmma_f32_16x16x32_f16(false, a, false, b,
                                                 (short)0, d, false, false);
    }

    // epilogue for this N tile: u = relu(d + b2[n]); acc += u * W3[n]
    const float b2v = b2[nb + col];
    const float w3v = W3[nb + col];
#pragma unroll
    for (int r = 0; r < 8; ++r) {
      float u = d[r] + b2v;
      u = u > 0.f ? u : 0.f;
      float t = u * w3v;
      t += __shfl_xor(t, 1, 32);       // reduce across the 16 columns of this half
      t += __shfl_xor(t, 2, 32);
      t += __shfl_xor(t, 4, 32);
      t += __shfl_xor(t, 8, 32);
      acc[r] += t;
    }
  }

  // ---- write: lane 0 holds rows 0..7, lane 16 holds rows 8..15 of the M-tile
  if (col == 0) {
    const float bias3 = b3[0];
    const int mbase = hi * 8;
#pragma unroll
    for (int r = 0; r < 8; ++r) {
      int m  = mbase + r;
      int p  = wave * 16 + m;
      int il = p >> 3, jl = p & 7;
      out[(size_t)(i0 + il) * BSZ + (j0 + jl)] = acc[r] + bias3;
    }
  }
}

// ----------------------------- launch --------------------------------------
extern "C" void kernel_launch(void* const* d_in, const int* in_sizes, int n_in,
                              void* d_out, int out_size, void* d_ws, size_t ws_size,
                              hipStream_t stream) {
  const float* x  = (const float*)d_in[0];
  const float* y  = (const float*)d_in[1];
  const float* W1 = (const float*)d_in[2];
  const float* b1 = (const float*)d_in[3];
  const float* W2 = (const float*)d_in[4];
  const float* b2 = (const float*)d_in[5];
  const float* W3 = (const float*)d_in[6];
  const float* b3 = (const float*)d_in[7];

  float*    hx  = (float*)d_ws;                 // 512*512 f32 = 1 MB
  float*    hy  = hx + BSZ * HID;               // 1 MB
  _Float16* W2h = (_Float16*)(hy + BSZ * HID);  // 512*512 f16 = 0.5 MB

  cc_prep<<<3072, 256, 0, stream>>>(x, y, W1, W2, hx, hy, W2h);

  dim3 grid(BSZ / 16, BSZ / 8);                 // 32 x 64 workgroups
  cc_main<<<grid, 256, 0, stream>>>(hx, hy, b1, W2h, b2, W3, b3, (float*)d_out);
}
